// MambaBlock_50500225466522
// MI455X (gfx1250) — matrix operand order repaired and checked by reading, not compiled, not run on hardware
//
#include <hip/hip_runtime.h>
#include <hip/hip_bf16.h>
#include <cstdint>

#define NUM_LAYERS 8
#define D_MODEL    512
#define D_INNER    1024
#define D_STATE    16
#define D_CONV     4
#define DT_RANK    32
#define BATCH      4
#define SEQ        1024
#define TOKENS     (BATCH * SEQ)   // 4096

typedef __bf16 bf16_t;
typedef bf16_t          v16bf __attribute__((ext_vector_type(16)));
typedef bf16_t          v8bf  __attribute__((ext_vector_type(8)));
typedef float           v8f   __attribute__((ext_vector_type(8)));

// 32-bit LDS byte-offset of a __shared__ object (AS3 pointers are 32-bit).
typedef const unsigned char __attribute__((address_space(3)))* lds_cptr_t;
__device__ __forceinline__ unsigned lds_offset(const void* p) {
    return (unsigned)(uintptr_t)(lds_cptr_t)p;
}

__device__ __forceinline__ float sigmoid_f(float x) { return 1.0f / (1.0f + __expf(-x)); }

// per-lane 16B global -> LDS DMA, tracked by ASYNCcnt
__device__ __forceinline__ void stage_async_b128(const void* gaddr, unsigned ldsoff) {
    unsigned long long g = (unsigned long long)(uintptr_t)gaddr;
    asm volatile("global_load_async_to_lds_b128 %0, %1, off" :: "v"(ldsoff), "v"(g) : "memory");
}
__device__ __forceinline__ void wait_async0() {
    asm volatile("s_wait_asynccnt 0x0" ::: "memory");
}

// hardware float -> bf16 conversion (v_cvt_pk_bf16_f32)
__device__ __forceinline__ v16bf cvt16(const float* f) {
    v16bf v;
    #pragma unroll
    for (int i = 0; i < 16; ++i) v[i] = (bf16_t)f[i];
    return v;
}

// -----------------------------------------------------------------------------
// Bulk f32 -> bf16 weight conversion (8 floats/thread, b128 in / b128-sized out)
// -----------------------------------------------------------------------------
__global__ __launch_bounds__(256)
void f32_to_bf16_kernel(const float* __restrict__ in, bf16_t* __restrict__ out) {
    const size_t i0 = ((size_t)blockIdx.x * 256 + threadIdx.x) * 8;
    float buf[8];
    *(float4*)(buf)     = *(const float4*)(in + i0);
    *(float4*)(buf + 4) = *(const float4*)(in + i0 + 4);
    v8bf o;
    #pragma unroll
    for (int i = 0; i < 8; ++i) o[i] = (bf16_t)buf[i];
    *(v8bf*)(out + i0) = o;
}

// -----------------------------------------------------------------------------
// C[M x N] = A[M x K (row stride lda)] * Wb[N x K]^T, bf16 WMMA, f32 accumulate.
// grid = (N/64, M/128), block = 256 (8 wave32). Each wave: 16 M-rows x 64 N-cols.
// Pre-converted bf16 weight tiles (64 x 32) double-buffered in LDS via
// global_load_async_to_lds_b128: DMA of tile k+1 overlaps WMMAs on tile k.
// epi==1: out = softplus(out + bias[col])   (dt_proj fused epilogue)
// -----------------------------------------------------------------------------
__global__ __launch_bounds__(256)
void gemm_bf16_wmma(const float* __restrict__ A, const bf16_t* __restrict__ W,
                    float* __restrict__ C, const float* __restrict__ bias,
                    int K, int N, int lda, int epi) {
    __shared__ __align__(16) bf16_t wt[2][64][40];   // 2 x (64 rows x 32 bf16 +16B pad)

    const int t    = threadIdx.x;
    const int wid  = t >> 5;
    const int lane = t & 31;
    const int r    = lane & 15;    // A: row within tile / B,C: column within tile
    const int h    = lane >> 4;    // half-wave select
    const int nbase = blockIdx.x * 64;
    const int mbase = blockIdx.y * 128 + wid * 16;

    // async staging: 64 rows x 32 bf16 = 256 x 16B; one b128 per thread
    const int srow = t >> 2;             // 0..63
    const int scg  = (t & 3) * 8;        // bf16 column within row
    const bf16_t* wrowp = W + (size_t)(nbase + srow) * K + scg;
    const unsigned lmine = lds_offset(&wt[0][srow][scg]);
    const unsigned lbufstride = (unsigned)sizeof(bf16_t) * 64 * 40;   // 5120 B

    v8f acc[4] = {};
    const float* arow = A + (size_t)(mbase + r) * lda;

    // prologue: stage first tile
    stage_async_b128(wrowp, lmine);
    wait_async0();
    __syncthreads();

    int cur = 0;
    for (int kb = 0; kb < K; kb += 32) {
        const bool more = (kb + 32 < K);
        if (more) {   // overlap DMA of next tile with compute on current tile
            stage_async_b128(wrowp + kb + 32, lmine + (cur ^ 1) * lbufstride);
            __builtin_prefetch(arow + kb + 32, 0, 1);
        }

        // A fragment: element e -> k = e + 16*h (matches B order)
        float abuf[16];
        {
            const float* ap = arow + kb;
            *(float4*)(abuf + 0)  = *(const float4*)(ap + h * 16);
            *(float4*)(abuf + 4)  = *(const float4*)(ap + h * 16 + 4);
            *(float4*)(abuf + 8)  = *(const float4*)(ap + h * 16 + 8);
            *(float4*)(abuf + 12) = *(const float4*)(ap + h * 16 + 12);
        }
        const v16bf afrag = cvt16(abuf);

        // B fragments: lane r = column n, element e -> k = e + 16*h (16 contiguous bf16)
        v16bf bfrag[4];
        #pragma unroll
        for (int nt = 0; nt < 4; ++nt) {
            const bf16_t* bp = &wt[cur][nt * 16 + r][h * 16];
            union { struct { v8bf lo, hi; } p; v16bf v; } u_;
            u_.p.lo = *(const v8bf*)(bp);
            u_.p.hi = *(const v8bf*)(bp + 8);
            bfrag[nt] = u_.v;
        }

        #pragma unroll
        for (int nt = 0; nt < 4; ++nt)
            acc[nt] = __builtin_amdgcn_wmma_f32_16x16x32_bf16(
                false, afrag, false, bfrag[nt], (short)0, acc[nt], false, false);

        if (more) {
            wait_async0();      // next tile resident
            __syncthreads();    // all waves done with cur + staging complete
        }
        cur ^= 1;
    }

    // C/D layout: VGPR j, lanes 0-15 -> M=j, N=lane; lanes 16-31 -> M=j+8
    #pragma unroll
    for (int nt = 0; nt < 4; ++nt) {
        const int col = nbase + nt * 16 + r;
        #pragma unroll
        for (int j = 0; j < 8; ++j) {
            const int row = mbase + j + 8 * h;
            float v = acc[nt][j];
            if (epi == 1) {
                v += bias[col];
                v = (v > 20.0f) ? v : logf(1.0f + __expf(v));   // softplus
            }
            C[(size_t)row * N + col] = v;
        }
    }
}

// -----------------------------------------------------------------------------
// Depthwise causal conv (k=4) + bias + SiLU.  xin = xz[:, 0:D_INNER].
// -----------------------------------------------------------------------------
__global__ __launch_bounds__(256)
void conv_silu_kernel(const float* __restrict__ xz, const float* __restrict__ cw,
                      const float* __restrict__ cb, float* __restrict__ u) {
    const int idx = blockIdx.x * 256 + threadIdx.x;     // token*D_INNER + d
    const int d   = idx & (D_INNER - 1);
    const int bl  = idx >> 10;                          // global token index b*SEQ + l
    const int l   = bl & (SEQ - 1);
    float acc = cb[d];
    #pragma unroll
    for (int j = 0; j < D_CONV; ++j) {
        const int ls = l + j - (D_CONV - 1);
        if (ls >= 0)
            acc += xz[(size_t)(bl + j - (D_CONV - 1)) * (2 * D_INNER) + d] * cw[d * D_CONV + j];
    }
    u[idx] = acc * sigmoid_f(acc);
}

// -----------------------------------------------------------------------------
// Selective scan: one thread per (b, d) channel, 16 states in registers,
// B/C vectors staged through LDS in 32-step chunks (shared across channels).
// Fused epilogue: yf = (y + u*D) * silu(z).
// -----------------------------------------------------------------------------
__global__ __launch_bounds__(256)
void scan_kernel(const float* __restrict__ dbc, const float* __restrict__ delta,
                 const float* __restrict__ u, const float* __restrict__ xz,
                 const float* __restrict__ Alog, const float* __restrict__ Dsk,
                 float* __restrict__ yf) {
    __shared__ float sBC[32][32];                       // [step][0:16]=B, [16:32]=C
    const int b = blockIdx.x >> 2;                      // D_INNER/256 = 4 chunks
    const int d = ((blockIdx.x & 3) << 8) + threadIdx.x;

    float A[D_STATE], h[D_STATE];
    #pragma unroll
    for (int s = 0; s < D_STATE; ++s) {
        A[s] = -__expf(Alog[d * D_STATE + s]);
        h[s] = 0.0f;
    }
    const float Dd = Dsk[d];

    for (int l0 = 0; l0 < SEQ; l0 += 32) {
        __syncthreads();
        #pragma unroll
        for (int q = 0; q < 4; ++q) {                   // 1024 floats, 256 threads
            const int e  = threadIdx.x + q * 256;
            const int rr = e >> 5, cc = e & 31;
            sBC[rr][cc] = dbc[(size_t)(b * SEQ + l0 + rr) * 64 + 32 + cc];
        }
        __syncthreads();
        for (int li = 0; li < 32; ++li) {
            const size_t off = (size_t)(b * SEQ + l0 + li) * D_INNER + d;
            const float dt = delta[off];
            const float uu = u[off];
            const float du = dt * uu;
            float y = 0.0f;
            #pragma unroll
            for (int s = 0; s < D_STATE; ++s) {
                const float dA = __expf(dt * A[s]);
                h[s] = dA * h[s] + du * sBC[li][s];
                y += h[s] * sBC[li][16 + s];
            }
            const float zz = xz[(size_t)(b * SEQ + l0 + li) * (2 * D_INNER) + D_INNER + d];
            yf[off] = (y + uu * Dd) * (zz * sigmoid_f(zz));
        }
    }
}

// -----------------------------------------------------------------------------
extern "C" void kernel_launch(void* const* d_in, const int* in_sizes, int n_in,
                              void* d_out, int out_size, void* d_ws, size_t ws_size,
                              hipStream_t stream) {
    (void)in_sizes; (void)n_in; (void)out_size; (void)ws_size;
    const float* x    = (const float*)d_in[0];
    const float* Wi   = (const float*)d_in[1];
    const float* cw   = (const float*)d_in[2];
    const float* cb   = (const float*)d_in[3];
    const float* Wx   = (const float*)d_in[4];
    const float* Wdt  = (const float*)d_in[5];
    const float* bdt  = (const float*)d_in[6];
    const float* Alog = (const float*)d_in[7];
    const float* Dsk  = (const float*)d_in[8];
    const float* Wo   = (const float*)d_in[9];
    float* out = (float*)d_out;

    float* w = (float*)d_ws;
    float* xz    = w; w += (size_t)TOKENS * 2 * D_INNER;   // 32 MB
    float* u     = w; w += (size_t)TOKENS * D_INNER;       // 16 MB
    float* dbc   = w; w += (size_t)TOKENS * 64;            //  1 MB
    float* delta = w; w += (size_t)TOKENS * D_INNER;       // 16 MB
    float* yf    = w; w += (size_t)TOKENS * D_INNER;       // 16 MB
    float* xbuf  = w; w += (size_t)TOKENS * D_MODEL;       //  8 MB

    // bf16 weight copies (converted once per launch)
    const size_t nWi  = (size_t)NUM_LAYERS * 2 * D_INNER * D_MODEL;   // 8.4 M
    const size_t nWx  = (size_t)NUM_LAYERS * 64 * D_INNER;            // 0.5 M
    const size_t nWdt = (size_t)NUM_LAYERS * D_INNER * DT_RANK;       // 0.26 M
    const size_t nWo  = (size_t)NUM_LAYERS * D_MODEL * D_INNER;       // 4.2 M
    bf16_t* wb = (bf16_t*)w;
    bf16_t* Wi_b  = wb; wb += nWi;
    bf16_t* Wx_b  = wb; wb += nWx;
    bf16_t* Wdt_b = wb; wb += nWdt;
    bf16_t* Wo_b  = wb; wb += nWo;

    f32_to_bf16_kernel<<<nWi  / 2048, 256, 0, stream>>>(Wi,  Wi_b);
    f32_to_bf16_kernel<<<nWx  / 2048, 256, 0, stream>>>(Wx,  Wx_b);
    f32_to_bf16_kernel<<<nWdt / 2048, 256, 0, stream>>>(Wdt, Wdt_b);
    f32_to_bf16_kernel<<<nWo  / 2048, 256, 0, stream>>>(Wo,  Wo_b);

    const float* xcur = x;
    for (int i = 0; i < NUM_LAYERS; ++i) {
        const bf16_t* Wi_i  = Wi_b  + (size_t)i * 2 * D_INNER * D_MODEL;
        const float*  cw_i  = cw    + (size_t)i * D_INNER * D_CONV;
        const float*  cb_i  = cb    + (size_t)i * D_INNER;
        const bf16_t* Wx_i  = Wx_b  + (size_t)i * 64 * D_INNER;
        const bf16_t* Wdt_i = Wdt_b + (size_t)i * D_INNER * DT_RANK;
        const float*  bdt_i = bdt   + (size_t)i * D_INNER;
        const float*  Al_i  = Alog  + (size_t)i * D_INNER * D_STATE;
        const float*  Dk_i  = Dsk   + (size_t)i * D_INNER;
        const bf16_t* Wo_i  = Wo_b  + (size_t)i * D_MODEL * D_INNER;

        // in_proj: xz[4096 x 2048] = x @ Wi^T   (K = 512)
        gemm_bf16_wmma<<<dim3(2 * D_INNER / 64, TOKENS / 128), 256, 0, stream>>>(
            xcur, Wi_i, xz, nullptr, D_MODEL, 2 * D_INNER, D_MODEL, 0);

        // depthwise causal conv + SiLU -> u[4096 x 1024]
        conv_silu_kernel<<<(TOKENS * D_INNER) / 256, 256, 0, stream>>>(xz, cw_i, cb_i, u);

        // x_proj: dbc[4096 x 64] = u @ Wx^T   (K = 1024)
        gemm_bf16_wmma<<<dim3(64 / 64, TOKENS / 128), 256, 0, stream>>>(
            u, Wx_i, dbc, nullptr, D_INNER, 64, D_INNER, 0);

        // dt_proj + softplus: delta[4096 x 1024] = softplus(dbc[:, :32] @ Wdt^T + bdt)
        gemm_bf16_wmma<<<dim3(D_INNER / 64, TOKENS / 128), 256, 0, stream>>>(
            dbc, Wdt_i, delta, bdt_i, DT_RANK, D_INNER, 64, 1);

        // selective scan + gating epilogue -> yf[4096 x 1024]
        scan_kernel<<<BATCH * (D_INNER / 256), 256, 0, stream>>>(
            dbc, delta, u, xz, Al_i, Dk_i, yf);

        // out_proj: x_next[4096 x 512] = yf @ Wo^T   (K = 1024)
        float* xdst = (i == NUM_LAYERS - 1) ? out : xbuf;
        gemm_bf16_wmma<<<dim3(D_MODEL / 64, TOKENS / 128), 256, 0, stream>>>(
            yf, Wo_i, xdst, nullptr, D_INNER, D_MODEL, D_INNER, 0);

        xcur = xbuf;
    }
}
